// VectorQuantizer_21792664060742
// MI455X (gfx1250) — matrix operand order repaired and checked by reading, not compiled, not run on hardware
//
#include <hip/hip_runtime.h>

// ---------------------------------------------------------------------------
// VQ-VAE encoder + vector quantizer for MI455X (gfx1250, wave32, WMMA).
// GEMMs on v_wmma_f32_16x16x32_bf16; tile staging via Tensor Data Mover
// (tensor_load_to_lds + s_wait_tensorcnt) with double-buffered LDS.
// ---------------------------------------------------------------------------

typedef __bf16 bf16_t;
typedef __bf16 v8bf  __attribute__((ext_vector_type(8)));
typedef __bf16 v16bf __attribute__((ext_vector_type(16)));
typedef float  v8f   __attribute__((ext_vector_type(8)));
typedef float  v4f   __attribute__((ext_vector_type(4)));
typedef unsigned int u32x4 __attribute__((ext_vector_type(4)));
typedef int          i32x4 __attribute__((ext_vector_type(4)));
typedef int          i32x8 __attribute__((ext_vector_type(8)));

#define N_TOK 16384
#define D_IN  512
#define HID   1024
#define D_EMB 256
#define K_CB  8192

#if defined(__has_builtin)
#if __has_builtin(__builtin_amdgcn_tensor_load_to_lds)
#define HAVE_TDM 1
#else
#define HAVE_TDM 0
#endif
#else
#define HAVE_TDM 0
#endif

// Build a 16-element bf16 fragment from two 16-byte chunks.
static __device__ __forceinline__ v16bf frag_ld(const bf16_t* p0, const bf16_t* p1) {
    v8bf lo = *(const v8bf*)p0;
    v8bf hi = *(const v8bf*)p1;
    return __builtin_shufflevector(lo, hi, 0,1,2,3,4,5,6,7,8,9,10,11,12,13,14,15);
}

static __device__ __forceinline__ unsigned lds_off_of(const void* p) {
    // Flat LDS-aperture address: low 32 bits are the workgroup-relative offset.
    return (unsigned)(unsigned long long)p;
}

#if HAVE_TDM
// Issue a 2D TDM load: tile_d1 rows of tile_d0 bf16 elements (row pitch
// stride_d0 elements) from global -> contiguous LDS at lds_off.
// D# packing per CDNA5 ISA (groups 0/1; 2/3 zero for 2D).
// This toolchain's builtin takes 6 args: (g0, g1, g2, g3, g4, cpol).
static __device__ __forceinline__ void tdm_load_2d(unsigned lds_off, const void* gaddr,
                                                   unsigned tile_d0, unsigned tile_d1,
                                                   unsigned tensor_d0, unsigned tensor_d1,
                                                   unsigned long long stride_d0) {
    unsigned long long ga = (unsigned long long)gaddr;
    u32x4 g0;
    g0.x = 1u;                                             // count=1 (valid), user mode
    g0.y = lds_off;                                        // lds_addr (bytes)
    g0.z = (unsigned)(ga & 0xFFFFFFFFull);                 // global_addr[31:0]
    g0.w = (unsigned)((ga >> 32) & 0x1FFFFFFull)           // global_addr[56:32]
           | (2u << 30);                                   // type = 2 ("image")
    i32x8 g1;
    g1[0] = (int)(1u << 16);                               // data_size=1 -> 2 bytes
    g1[1] = (int)((tensor_d0 & 0xFFFFu) << 16);            // tensor_dim0[15:0]
    g1[2] = (int)(((tensor_d0 >> 16) & 0xFFFFu)            // tensor_dim0[31:16]
                  | ((tensor_d1 & 0xFFFFu) << 16));        // tensor_dim1[15:0]
    g1[3] = (int)(((tensor_d1 >> 16) & 0xFFFFu)            // tensor_dim1[31:16]
                  | ((tile_d0 & 0xFFFFu) << 16));          // tile_dim0
    g1[4] = (int)(tile_d1 & 0xFFFFu);                      // tile_dim1 (tile_dim2=0)
    g1[5] = (int)(stride_d0 & 0xFFFFFFFFull);              // tensor_dim0_stride[31:0]
    g1[6] = (int)((stride_d0 >> 32) & 0xFFFFull);          // tensor_dim0_stride[47:32]
    g1[7] = 0;
    i32x4 z4 = {0, 0, 0, 0};
    i32x8 z8 = {0, 0, 0, 0, 0, 0, 0, 0};
    __builtin_amdgcn_tensor_load_to_lds(g0, g1, z4, z4, z8, 0);
}
#endif

// ---------------------------------------------------------------------------
// Prep kernels
// ---------------------------------------------------------------------------
__global__ void cvt_f32_bf16(const float* __restrict__ in, bf16_t* __restrict__ out, int n) {
    int i = blockIdx.x * blockDim.x + threadIdx.x;
    int stride = gridDim.x * blockDim.x;
    for (; i < n; i += stride) out[i] = (bf16_t)in[i];
}

// W is [din, dout] row-major (as in x @ W). Produce Wt [dout, din] bf16 (K-contiguous).
__global__ void transpose_cvt(const float* __restrict__ W, bf16_t* __restrict__ Wt,
                              int din, int dout) {
    int t = blockIdx.x * blockDim.x + threadIdx.x;
    int total = din * dout;
    int stride = gridDim.x * blockDim.x;
    for (; t < total; t += stride) {
        int o = t / din;
        int i = t - o * din;
        Wt[t] = (bf16_t)W[(size_t)i * dout + o];   // t == o*din + i
    }
}

// One wave per codebook row: cast to bf16 and compute ||e||^2.
__global__ void emb_prep(const float* __restrict__ emb, bf16_t* __restrict__ emb_bf,
                         float* __restrict__ enorm2) {
    const int wave = threadIdx.x >> 5;
    const int lane = threadIdx.x & 31;
    const int row  = blockIdx.x * 8 + wave;          // grid = K_CB/8 blocks
    const float* src = emb + (size_t)row * D_EMB + lane * 8;
    bf16_t*      dst = emb_bf + (size_t)row * D_EMB + lane * 8;
    float s = 0.f;
#pragma unroll
    for (int j = 0; j < 8; ++j) { float v = src[j]; s += v * v; dst[j] = (bf16_t)v; }
#pragma unroll
    for (int off = 16; off >= 1; off >>= 1) s += __shfl_xor(s, off, 32);
    if (lane == 0) enorm2[row] = s;
}

// ---------------------------------------------------------------------------
// bf16 WMMA GEMM:  C[M,N] = act(A[M,Kd] * Bt[N,Kd]^T + bias)
// Block = 256 threads (8 waves): 128 M-rows (16 per wave) x 64 N-cols.
// B tile (64x32 bf16 = 4 KB) staged into double-buffered LDS by the TDM;
// wave 0 issues the descriptor one K-step ahead and waits on TENSORcnt.
// ---------------------------------------------------------------------------
__global__ __launch_bounds__(256, 2)
void gemm_bf16_wmma(const bf16_t* __restrict__ A, const bf16_t* __restrict__ Bt,
                    const float* __restrict__ bias, bf16_t* __restrict__ C,
                    int M, int N, int Kd, int relu) {
    __shared__ bf16_t bsh[2][64][32];

    const int wave = threadIdx.x >> 5;
    const int lane = threadIdx.x & 31;
    const int n0   = blockIdx.x * 64;
    const int m0   = blockIdx.y * 128;
    const int mrow = m0 + wave * 16 + (lane & 15);
    const int ka   = (lane >> 4) * 8;    // A frag: lanes 0-15 -> K 0-7/16-23, 16-31 -> +8
    const int kb   = (lane >> 4) * 8;    // B frag: lanes 0-15 -> K 0-7/16-23, 16-31 -> +8
    const int nK   = Kd >> 5;

    v8f acc[4] = {};

#if HAVE_TDM
    const unsigned lds0 = lds_off_of(&bsh[0][0][0]);
    const bf16_t* gB = Bt + (size_t)n0 * Kd;
    if (threadIdx.x < 32)
        tdm_load_2d(lds0, gB, 32, 64, 32, 64, (unsigned long long)Kd);
#endif

    for (int ks = 0; ks < nK; ++ks) {
        const int k0 = ks << 5;
#if HAVE_TDM
        const int buf = ks & 1;
        if (threadIdx.x < 32) {
            if (ks + 1 < nK) {
                tdm_load_2d(lds0 + (unsigned)((buf ^ 1) * 4096),
                            gB + ((size_t)(ks + 1) << 5), 32, 64, 32, 64,
                            (unsigned long long)Kd);
                __builtin_amdgcn_s_wait_tensorcnt(1);   // current buffer ready
            } else {
                __builtin_amdgcn_s_wait_tensorcnt(0);
            }
        }
        __syncthreads();
        const bf16_t (*bt)[32] = bsh[buf];
#else
        const int ldr = threadIdx.x >> 2;        // 0..63 : B tile row
        const int ldc = (threadIdx.x & 3) * 8;   // 4 x 16B chunks per row
        *(v8bf*)&bsh[0][ldr][ldc] = *(const v8bf*)&Bt[(size_t)(n0 + ldr) * Kd + k0 + ldc];
        __syncthreads();
        const bf16_t (*bt)[32] = bsh[0];
#endif
        v16bf af = frag_ld(&A[(size_t)mrow * Kd + k0 + ka],
                           &A[(size_t)mrow * Kd + k0 + ka + 16]);
#pragma unroll
        for (int j = 0; j < 4; ++j) {
            v16bf bfr = frag_ld(&bt[j * 16 + (lane & 15)][kb],
                                &bt[j * 16 + (lane & 15)][kb + 16]);
            acc[j] = __builtin_amdgcn_wmma_f32_16x16x32_bf16(
                         false, af, false, bfr, (short)0, acc[j], false, false);
        }
        __syncthreads();   // all reads done before this buffer is refilled
    }

    // Epilogue: C-tile lane l holds (m = r + 8*(l>>4), n = l&15) in VGPR r.
    const int mhalf = (lane >> 4) * 8;
#pragma unroll
    for (int j = 0; j < 4; ++j) {
        const int col = n0 + j * 16 + (lane & 15);
        const float bv = bias[col];
#pragma unroll
        for (int r = 0; r < 8; ++r) {
            float v = acc[j][r] + bv;
            if (relu) v = fmaxf(v, 0.f);
            C[(size_t)(m0 + wave * 16 + mhalf + r) * N + col] = (bf16_t)v;
        }
    }
}

// ---------------------------------------------------------------------------
// Distance + argmin: per wave, 16 tokens held fully in registers (8 A frags,
// D_EMB=256). Codebook streamed in 16-row tiles (8 KB) via TDM into a
// double-buffered LDS; score = ||e||^2 - 2 z.e ; per-lane running argmin,
// then a 16-lane shfl_xor argmin reduction.
// ---------------------------------------------------------------------------
__global__ __launch_bounds__(256, 2)
void vq_argmin_wmma(const bf16_t* __restrict__ Z, const bf16_t* __restrict__ Ebf,
                    const float* __restrict__ enorm2, int* __restrict__ idx_out) {
    __shared__ bf16_t esh[2][16][D_EMB];  // 2 x 8 KB codebook tiles

    const int wave = threadIdx.x >> 5;
    const int lane = threadIdx.x & 31;
    const int tok0 = blockIdx.x * 128;
    const int myrow = tok0 + wave * 16 + (lane & 15);
    const int ka = (lane >> 4) * 8;
    const int kb = (lane >> 4) * 8;

    v16bf afrag[8];
#pragma unroll
    for (int s = 0; s < 8; ++s)
        afrag[s] = frag_ld(&Z[(size_t)myrow * D_EMB + s * 32 + ka],
                           &Z[(size_t)myrow * D_EMB + s * 32 + ka + 16]);

    float minv[8];
    int   mini[8];
#pragma unroll
    for (int r = 0; r < 8; ++r) { minv[r] = 3.402823466e38f; mini[r] = 0; }

#if HAVE_TDM
    const unsigned lds0 = lds_off_of(&esh[0][0][0]);
    if (threadIdx.x < 32)
        tdm_load_2d(lds0, Ebf, D_EMB, 16, D_EMB, 16, D_EMB);
#endif

    const int ntiles = K_CB / 16;
    for (int t = 0; t < ntiles; ++t) {
        const int c0 = t << 4;
#if HAVE_TDM
        const int buf = t & 1;
        if (threadIdx.x < 32) {
            if (t + 1 < ntiles) {
                tdm_load_2d(lds0 + (unsigned)((buf ^ 1) * 8192),
                            Ebf + (size_t)(c0 + 16) * D_EMB, D_EMB, 16, D_EMB, 16, D_EMB);
                __builtin_amdgcn_s_wait_tensorcnt(1);
            } else {
                __builtin_amdgcn_s_wait_tensorcnt(0);
            }
        }
        __syncthreads();
        const bf16_t (*eb)[D_EMB] = esh[buf];
#else
        const int ldr = threadIdx.x >> 4;         // 0..15 : codebook row in tile
        const int ldc = (threadIdx.x & 15) * 16;  // 16 x 32B per row
        *(v8bf*)&esh[0][ldr][ldc]     = *(const v8bf*)&Ebf[(size_t)(c0 + ldr) * D_EMB + ldc];
        *(v8bf*)&esh[0][ldr][ldc + 8] = *(const v8bf*)&Ebf[(size_t)(c0 + ldr) * D_EMB + ldc + 8];
        __syncthreads();
        const bf16_t (*eb)[D_EMB] = esh[0];
#endif
        v8f acc = {};
#pragma unroll
        for (int s = 0; s < 8; ++s) {
            v16bf bfr = frag_ld(&eb[lane & 15][s * 32 + kb],
                                &eb[lane & 15][s * 32 + kb + 16]);
            acc = __builtin_amdgcn_wmma_f32_16x16x32_bf16(
                      false, afrag[s], false, bfr, (short)0, acc, false, false);
        }

        const float en   = enorm2[c0 + (lane & 15)];   // L2-hot, 1 dword/lane/tile
        const int   ncol = c0 + (lane & 15);
#pragma unroll
        for (int r = 0; r < 8; ++r) {
            float score = en - 2.f * acc[r];
            if (score < minv[r]) { minv[r] = score; mini[r] = ncol; }
        }
        __syncthreads();   // all reads done before this buffer is refilled
    }

    // Argmin across the 16 lanes that share one M row (xor of bits 0..3 only,
    // never crossing the lane-16 boundary that separates M=r from M=r+8).
#pragma unroll
    for (int r = 0; r < 8; ++r) {
#pragma unroll
        for (int off = 8; off >= 1; off >>= 1) {
            float ov = __shfl_xor(minv[r], off, 32);
            int   oi = __shfl_xor(mini[r], off, 32);
            if (ov < minv[r] || (ov == minv[r] && oi < mini[r])) {
                minv[r] = ov; mini[r] = oi;
            }
        }
        if ((lane & 15) == 0) {
            const int m = (lane >> 4) * 8 + r;
            idx_out[tok0 + wave * 16 + m] = mini[r];
        }
    }
}

// ---------------------------------------------------------------------------
// out[n, :] = emb_f32[idx[n], :]   (one block per token, 64 threads x float4)
// ---------------------------------------------------------------------------
__global__ void gather_rows(const float* __restrict__ emb, const int* __restrict__ idx,
                            float* __restrict__ out) {
    const int t = blockIdx.x;
    const int code = idx[t];
    const v4f* src = (const v4f*)(emb + (size_t)code * D_EMB);
    v4f*       dst = (v4f*)(out + (size_t)t * D_EMB);
    dst[threadIdx.x] = src[threadIdx.x];
}

// ---------------------------------------------------------------------------
// Host launch
// ---------------------------------------------------------------------------
extern "C" void kernel_launch(void* const* d_in, const int* in_sizes, int n_in,
                              void* d_out, int out_size, void* d_ws, size_t ws_size,
                              hipStream_t stream) {
    (void)in_sizes; (void)n_in; (void)out_size; (void)ws_size;

    const float* x   = (const float*)d_in[0];
    const float* W0  = (const float*)d_in[1];
    const float* b0  = (const float*)d_in[2];
    const float* W1  = (const float*)d_in[3];
    const float* b1  = (const float*)d_in[4];
    const float* W2  = (const float*)d_in[5];
    const float* b2  = (const float*)d_in[6];
    const float* emb = (const float*)d_in[7];
    float* out = (float*)d_out;

    char* ws = (char*)d_ws;
    size_t off = 0;
    auto take = [&](size_t bytes) -> void* {
        void* p = ws + off;
        off += (bytes + 255) & ~(size_t)255;
        return p;
    };

    bf16_t* x_bf   = (bf16_t*)take((size_t)N_TOK * D_IN  * 2);  // 16 MB
    bf16_t* h0_bf  = (bf16_t*)take((size_t)N_TOK * HID   * 2);  // 32 MB
    bf16_t* h1_bf  = (bf16_t*)take((size_t)N_TOK * HID   * 2);  // 32 MB
    bf16_t* z_bf   = (bf16_t*)take((size_t)N_TOK * D_EMB * 2);  //  8 MB
    bf16_t* wt0    = (bf16_t*)take((size_t)HID   * D_IN  * 2);  //  1 MB
    bf16_t* wt1    = (bf16_t*)take((size_t)HID   * HID   * 2);  //  2 MB
    bf16_t* wt2    = (bf16_t*)take((size_t)D_EMB * HID   * 2);  // 0.5 MB
    bf16_t* emb_bf = (bf16_t*)take((size_t)K_CB  * D_EMB * 2);  //  4 MB
    float*  enorm2 = (float*) take((size_t)K_CB * 4);
    int*    idx    = (int*)   take((size_t)N_TOK * 4);

    // --- prep ---
    cvt_f32_bf16<<<4096, 256, 0, stream>>>(x, x_bf, N_TOK * D_IN);
    transpose_cvt<<<2048, 256, 0, stream>>>(W0, wt0, D_IN, HID);
    transpose_cvt<<<2048, 256, 0, stream>>>(W1, wt1, HID, HID);
    transpose_cvt<<<1024, 256, 0, stream>>>(W2, wt2, HID, D_EMB);
    emb_prep<<<K_CB / 8, 256, 0, stream>>>(emb, emb_bf, enorm2);

    // --- MLP encoder (all WMMA bf16, f32 accumulate) ---
    gemm_bf16_wmma<<<dim3(HID / 64,   N_TOK / 128), 256, 0, stream>>>(
        x_bf, wt0, b0, h0_bf, N_TOK, HID, D_IN, 1);
    gemm_bf16_wmma<<<dim3(HID / 64,   N_TOK / 128), 256, 0, stream>>>(
        h0_bf, wt1, b1, h1_bf, N_TOK, HID, HID, 1);
    gemm_bf16_wmma<<<dim3(D_EMB / 64, N_TOK / 128), 256, 0, stream>>>(
        h1_bf, wt2, b2, z_bf, N_TOK, D_EMB, HID, 0);

    // --- nearest codebook entry (dominant GEMM, fused argmin) ---
    vq_argmin_wmma<<<N_TOK / 128, 256, 0, stream>>>(z_bf, emb_bf, enorm2, idx);

    // --- straight-through forward value: quantized rows in fp32 ---
    gather_rows<<<N_TOK, 64, 0, stream>>>(emb, idx, out);
}
